// DenseExpert_678604833005
// MI455X (gfx1250) — matrix operand order repaired and checked by reading, not compiled, not run on hardware
//
#include <hip/hip_runtime.h>

// ---------------------------------------------------------------------------
// Gated dense-expert layer as ONE f16-WMMA GEMM on gfx1250 (wave32),
// now with double-buffered LDS software pipeline.
//   r[b,u] = elu( sum_e sum_d (g[b,e]*x[b,d]) * alpha[e,u,d]
//               + sum_e g[b,e]*beta[e,u] )
// A = gated x  [M=4096, K=E*D=4096] (gate + f32->f16 applied during staging)
// B = alpha    [N=512,  K=4096]     (alpha[E,U,D] is K-contiguous per (e,u))
// ---------------------------------------------------------------------------

typedef __attribute__((ext_vector_type(16))) _Float16 v16h;
typedef __attribute__((ext_vector_type(8)))  _Float16 v8h;
typedef __attribute__((ext_vector_type(8)))  float    v8f;

#define BSZ 4096
#define DD  512
#define UU  512
#define EE  8

#define BM  128      // block tile M
#define BN  128      // block tile N
#define KT  64       // K slab per LDS stage (2 wmma k-steps)
#define LDH 72       // LDS row stride in halfs (64 + 8 pad -> conflict-free)
#define NSLAB ((EE * DD) / KT)   // 64 pipeline stages

union HalfPack { v16h v; v8h h[2]; };

// ---- pipeline helpers -----------------------------------------------------

__device__ __forceinline__
void load_slab(const float* __restrict__ xrow, const float* __restrict__ ap,
               int urow, int s, int sc0, float4* ra, float4* rb)
{
    const int e  = s >> 3;          // expert index (8 slabs per expert)
    const int d0 = (s & 7) * KT;    // d offset within expert
    const float* xr = xrow + d0 + sc0;
    const float* ar = ap + ((size_t)e * UU + urow) * DD + d0 + sc0;
    #pragma unroll
    for (int j = 0; j < 8; ++j) ra[j] = *(const float4*)(xr + j * 4);
    #pragma unroll
    for (int j = 0; j < 8; ++j) rb[j] = *(const float4*)(ar + j * 4);
}

__device__ __forceinline__
void cvt_store(_Float16* Asb, _Float16* Bsb, int srow, int sc0, float gr,
               const float4* ra, const float4* rb)
{
    #pragma unroll
    for (int j = 0; j < 4; ++j) {
        const float4 f0 = ra[2 * j], f1 = ra[2 * j + 1];
        v8h h;
        h[0] = (_Float16)(gr * f0.x); h[1] = (_Float16)(gr * f0.y);
        h[2] = (_Float16)(gr * f0.z); h[3] = (_Float16)(gr * f0.w);
        h[4] = (_Float16)(gr * f1.x); h[5] = (_Float16)(gr * f1.y);
        h[6] = (_Float16)(gr * f1.z); h[7] = (_Float16)(gr * f1.w);
        *(v8h*)(Asb + srow * LDH + sc0 + j * 8) = h;
    }
    #pragma unroll
    for (int j = 0; j < 4; ++j) {
        const float4 f0 = rb[2 * j], f1 = rb[2 * j + 1];
        v8h h;
        h[0] = (_Float16)f0.x; h[1] = (_Float16)f0.y;
        h[2] = (_Float16)f0.z; h[3] = (_Float16)f0.w;
        h[4] = (_Float16)f1.x; h[5] = (_Float16)f1.y;
        h[6] = (_Float16)f1.z; h[7] = (_Float16)f1.w;
        *(v8h*)(Bsb + srow * LDH + sc0 + j * 8) = h;
    }
}

__device__ __forceinline__
void compute_slab(const _Float16* Asb, const _Float16* Bsb,
                  int wm, int wn, int lrow, int lhalf, v8f (&acc)[2][4])
{
    #pragma unroll
    for (int ks = 0; ks < 2; ++ks) {
        const int k0 = ks * 32;
        // A fragments: lane m holds K[{0..7,16..23}] (+8 for lanes>=16)
        HalfPack af[2];
        #pragma unroll
        for (int mi = 0; mi < 2; ++mi) {
            const _Float16* p =
                Asb + (wm * 32 + mi * 16 + lrow) * LDH + k0 + lhalf * 8;
            af[mi].h[0] = *(const v8h*)(p);
            af[mi].h[1] = *(const v8h*)(p + 16);
        }
        // B fragments: lane n holds K[{0..7,16..23}] (+8 for lanes>=16)
        HalfPack bf[4];
        #pragma unroll
        for (int ni = 0; ni < 4; ++ni) {
            const _Float16* p =
                Bsb + (wn * 64 + ni * 16 + lrow) * LDH + k0 + lhalf * 8;
            bf[ni].h[0] = *(const v8h*)(p);
            bf[ni].h[1] = *(const v8h*)(p + 16);
        }
        #pragma unroll
        for (int mi = 0; mi < 2; ++mi)
            #pragma unroll
            for (int ni = 0; ni < 4; ++ni)
                acc[mi][ni] = __builtin_amdgcn_wmma_f32_16x16x32_f16(
                    false, af[mi].v, false, bf[ni].v,
                    (short)0, acc[mi][ni], false, false);
    }
}

// ---- main kernel ----------------------------------------------------------

__global__ __launch_bounds__(256)
void moe_dense_wmma(const float* __restrict__ xp,     // [B,D]
                    const float* __restrict__ gp,     // [B,E]
                    const float* __restrict__ ap,     // [E,U,D]
                    const float* __restrict__ bp,     // [E,U]
                    float* __restrict__ op)           // [B,U]
{
    __shared__ _Float16 As[2][BM * LDH];
    __shared__ _Float16 Bs[2][BN * LDH];

    const int tid  = threadIdx.x;
    const int lane = tid & 31;
    const int wave = tid >> 5;          // 0..7
    const int wm   = wave & 3;          // wave tile row  (x32)
    const int wn   = wave >> 2;         // wave tile col  (x64)

    const int bm = blockIdx.y;          // 0..31  (M tiles)
    const int bn = blockIdx.x;          // 0..3   (N tiles)

    // staging assignment: 2 threads per row, 32 cols each
    const int srow = tid >> 1;          // 0..127
    const int sc0  = (tid & 1) * 32;    // 0 or 32

    const int grow = bm * BM + srow;    // global x row staged by this thread
    const int urow = bn * BN + srow;    // global alpha u-row staged by this thread

    const float* xrow = xp + (size_t)grow * DD;

    const int lhalf = (lane >> 4) & 1;  // lane-group selector for frag layout
    const int lrow  = lane & 15;

    v8f acc[2][4];
    #pragma unroll
    for (int mi = 0; mi < 2; ++mi)
        #pragma unroll
        for (int ni = 0; ni < 4; ++ni)
            #pragma unroll
            for (int q = 0; q < 8; ++q)
                acc[mi][ni][q] = 0.0f;

    // ---- software pipeline over 64 K-slabs, double-buffered LDS ----
    {
        // prologue: stage slab 0 into buffer 0
        float4 ra[8], rb[8];
        load_slab(xrow, ap, urow, 0, sc0, ra, rb);
        cvt_store(&As[0][0], &Bs[0][0], srow, sc0, gp[grow * EE + 0], ra, rb);
    }

    for (int s = 0; s < NSLAB; ++s) {
        const int buf = s & 1;
        float4 ra[8], rb[8];
        float grn = 0.0f;
        const bool has_next = (s + 1 < NSLAB);
        if (has_next) {
            grn = gp[grow * EE + ((s + 1) >> 3)];
            load_slab(xrow, ap, urow, s + 1, sc0, ra, rb);   // long-latency, early
        }
        __syncthreads();   // slab s visible; prior readers of buf^1 are done
        compute_slab(&As[buf][0], &Bs[buf][0], wm, wn, lrow, lhalf, acc);
        if (has_next)
            cvt_store(&As[buf ^ 1][0], &Bs[buf ^ 1][0], srow, sc0, grn, ra, rb);
    }

    // ---- epilogue: + sum_e g[b,e]*beta[e,u], then ELU ----
    const int colbase = bn * BN + wn * 64;
    float betav[4][8];
    #pragma unroll
    for (int ni = 0; ni < 4; ++ni) {
        const int u = colbase + ni * 16 + lrow;
        #pragma unroll
        for (int e = 0; e < EE; ++e)
            betav[ni][e] = bp[e * UU + u];
    }

    const int rowbase = bm * BM + wm * 32 + lhalf * 8;
    #pragma unroll
    for (int mi = 0; mi < 2; ++mi) {
        #pragma unroll
        for (int i = 0; i < 8; ++i) {
            const int row = rowbase + mi * 16 + i;
            const float4 g0 = *(const float4*)(gp + row * EE);
            const float4 g1 = *(const float4*)(gp + row * EE + 4);
            const float gv[8] = { g0.x, g0.y, g0.z, g0.w,
                                  g1.x, g1.y, g1.z, g1.w };
            #pragma unroll
            for (int ni = 0; ni < 4; ++ni) {
                float r = acc[mi][ni][i];
                #pragma unroll
                for (int e = 0; e < EE; ++e)
                    r += gv[e] * betav[ni][e];
                r = (r > 0.0f) ? r : (__expf(r) - 1.0f);   // elu
                op[(size_t)row * UU + colbase + ni * 16 + lrow] = r;
            }
        }
    }
}

extern "C" void kernel_launch(void* const* d_in, const int* in_sizes, int n_in,
                              void* d_out, int out_size, void* d_ws, size_t ws_size,
                              hipStream_t stream)
{
    const float* x     = (const float*)d_in[0];   // [4096,512]
    const float* g     = (const float*)d_in[1];   // [4096,8]
    const float* alpha = (const float*)d_in[2];   // [8,512,512]
    const float* beta  = (const float*)d_in[3];   // [8,512]
    float* out = (float*)d_out;                   // [4096,512]

    dim3 grid(UU / BN, BSZ / BM);                 // (4, 32)
    dim3 block(256);
    hipLaunchKernelGGL(moe_dense_wmma, grid, block, 0, stream,
                       x, g, alpha, beta, out);
}
